// Transformer_48799418417203
// MI455X (gfx1250) — matrix operand order repaired and checked by reading, not compiled, not run on hardware
//
#include <hip/hip_runtime.h>
#include <hip/hip_bf16.h>

typedef __bf16 bf16_t;
typedef __attribute__((ext_vector_type(16))) __bf16 v16bf;
typedef __attribute__((ext_vector_type(8)))  __bf16 v8bf;
typedef __attribute__((ext_vector_type(8)))  float  v8f;

union FragU { v16bf v; v8bf h[2]; };

#if defined(__HIP_DEVICE_COMPILE__) && __has_builtin(__builtin_amdgcn_tensor_load_to_lds) && __has_builtin(__builtin_amdgcn_s_wait_tensorcnt)
#define HAVE_TDM 1
#else
#define HAVE_TDM 0
#endif

__device__ inline v8f zero8() {
  v8f z;
#pragma unroll
  for (int i = 0; i < 8; ++i) z[i] = 0.0f;
  return z;
}

__device__ inline float wave_sum16(float v) {
  v += __shfl_xor(v, 1, 32); v += __shfl_xor(v, 2, 32);
  v += __shfl_xor(v, 4, 32); v += __shfl_xor(v, 8, 32);
  return v;
}
__device__ inline float wave_max16(float v) {
  v = fmaxf(v, __shfl_xor(v, 1, 32)); v = fmaxf(v, __shfl_xor(v, 2, 32));
  v = fmaxf(v, __shfl_xor(v, 4, 32)); v = fmaxf(v, __shfl_xor(v, 8, 32));
  return v;
}

#if HAVE_TDM
typedef __attribute__((ext_vector_type(4))) unsigned int u32x4;
typedef __attribute__((ext_vector_type(8))) int i32x8;
typedef __attribute__((ext_vector_type(4))) int i32x4;

// TDM: DMA a [tile1=128 rows x tile0=32 elem] bf16 tile (row stride = rowStride
// elems) from global into LDS, padding each 64B row with 16B -> LDS stride 40
// bf16 (conflict-free for the WMMA fragment b128 reads).
__device__ inline void tdm_load_tile(const bf16_t* gsrc, const bf16_t* ldst,
                                     int rowStride) {
  unsigned lds = (unsigned)(size_t)ldst;                 // low 32b = LDS offset
  unsigned long long ga = (unsigned long long)(size_t)gsrc;
  u32x4 g0;
  g0[0] = 1u;                                            // count=1, user desc
  g0[1] = lds;                                           // lds_addr
  g0[2] = (unsigned)ga;                                  // global_addr[31:0]
  g0[3] = (unsigned)((ga >> 32) & 0x01ffffffull) | 0x80000000u;  // [56:32]|type=2
  i32x8 g1;
  // data_size=2B (code 1) | pad_enable | pad_interval=16dw (3) | pad_amount=4dw (3)
  g1[0] = (int)((1u << 16) | (1u << 20) | (3u << 22) | (3u << 25));
  g1[1] = (int)(((unsigned)rowStride & 0xffffu) << 16);  // tensor_dim0[15:0]
  g1[2] = (int)(((unsigned)rowStride >> 16) | (128u << 16)); // dim0 hi | tensor_dim1
  g1[3] = (int)(32u << 16);                              // tile_dim0 = 32
  g1[4] = 128;                                           // tile_dim1 = 128
  g1[5] = rowStride;                                     // tensor_dim0_stride lo
  g1[6] = 0;
  g1[7] = 0;
  i32x4 z; z[0] = z[1] = z[2] = z[3] = 0;
#if defined(__clang_major__) && __clang_major__ >= 23
  i32x8 z8;
#pragma unroll
  for (int i = 0; i < 8; ++i) z8[i] = 0;
  __builtin_amdgcn_tensor_load_to_lds(g0, g1, z, z, z8, 0);
#else
  __builtin_amdgcn_tensor_load_to_lds(g0, g1, z, z, 0);
#endif
}
#endif  // HAVE_TDM

// ---------------------------------------------------------------------------
// Fold LoRA into QKV weight (bf16 out): Wm[e][d] = W[e][d] + sum_r B[e][r]*A[r][d]
// ---------------------------------------------------------------------------
__global__ __launch_bounds__(256)
void merge_qkv_kernel(const float* __restrict__ W, const float* __restrict__ A,
                      const float* __restrict__ Bl, bf16_t* __restrict__ Wm) {
  int idx = blockIdx.x * 256 + threadIdx.x;     // < 2304*768
  int e = idx / 768, d = idx % 768;
  float acc = W[idx];
#pragma unroll
  for (int r = 0; r < 8; ++r) acc += Bl[e * 8 + r] * A[r * 768 + d];
  Wm[idx] = (bf16_t)acc;
}

// f32 -> bf16 elementwise (weight pre-conversion)
__global__ __launch_bounds__(256)
void cvt_bf16_kernel(const float* __restrict__ in, bf16_t* __restrict__ out, int n) {
  int i = blockIdx.x * 256 + threadIdx.x;
  if (i < n) out[i] = (bf16_t)in[i];
}

// ---------------------------------------------------------------------------
// LayerNorm: one block per token row (768), output bf16
// ---------------------------------------------------------------------------
__global__ __launch_bounds__(256)
void ln_kernel(const float* __restrict__ x, const float* __restrict__ g,
               const float* __restrict__ b, bf16_t* __restrict__ h) {
  __shared__ float sred[8], ssred[8];
  const int row = blockIdx.x, tid = threadIdx.x;
  const float* xr = x + (size_t)row * 768;
  float vals[3], s = 0.f, ss = 0.f;
#pragma unroll
  for (int i = 0; i < 3; ++i) {
    float v = xr[tid + i * 256];
    vals[i] = v; s += v; ss += v * v;
  }
#pragma unroll
  for (int o = 16; o; o >>= 1) { s += __shfl_xor(s, o, 32); ss += __shfl_xor(ss, o, 32); }
  if ((tid & 31) == 0) { sred[tid >> 5] = s; ssred[tid >> 5] = ss; }
  __syncthreads();
  s = 0.f; ss = 0.f;
#pragma unroll
  for (int i = 0; i < 8; ++i) { s += sred[i]; ss += ssred[i]; }
  const float mu = s * (1.0f / 768.0f);
  const float var = ss * (1.0f / 768.0f) - mu * mu;
  const float r = rsqrtf(var + 1e-5f);
#pragma unroll
  for (int i = 0; i < 3; ++i) {
    int col = tid + i * 256;
    h[(size_t)row * 768 + col] = (bf16_t)((vals[i] - mu) * r * g[col] + b[col]);
  }
}

// ---------------------------------------------------------------------------
// Tiled bf16 WMMA GEMM: out[M,N] = A[M,K](bf16) * W[N,K]^T(bf16) (+bias)(+res)(gelu)
// TDM path: per 32-K tile one tensor_load_to_lds per matrix, double-buffered.
// EPI: 1=+bias  2=+bias+residual  3=+bias+quickGELU
// ---------------------------------------------------------------------------
#define LSTR 40   // LDS row stride in bf16 (80B rows -> conflict-free b128 reads)

template <int EPI, bool OUT_BF16>
__global__ __launch_bounds__(256)
void gemm_kernel(const bf16_t* __restrict__ A, const bf16_t* __restrict__ W,
                 const float* __restrict__ bias, const float* __restrict__ res,
                 void* __restrict__ out, int M, int N, int K) {
#if HAVE_TDM
  __shared__ bf16_t As[2][128 * LSTR];
  __shared__ bf16_t Ws[2][128 * LSTR];
#else
  __shared__ bf16_t As[1][128 * LSTR];
  __shared__ bf16_t Ws[1][128 * LSTR];
#endif
  const int tid = threadIdx.x, wid = tid >> 5, lane = tid & 31, l15 = lane & 15;
  const bool hi = lane >= 16;
  const int wm = wid >> 1, wn = wid & 1;          // 4x2 wave grid: 32 rows x 64 cols each
  const int bm = blockIdx.y * 128, bn = blockIdx.x * 128;
  const int steps = K >> 5;

  v8f acc[2][4];
#pragma unroll
  for (int i = 0; i < 2; ++i)
#pragma unroll
    for (int j = 0; j < 4; ++j) acc[i][j] = zero8();

#if HAVE_TDM
  if (wid == 0) {
    tdm_load_tile(A + (size_t)bm * K, &As[0][0], K);
    tdm_load_tile(W + (size_t)bn * K, &Ws[0][0], K);
  }
#endif

  for (int it = 0; it < steps; ++it) {
#if HAVE_TDM
    const int cur = it & 1;
    __syncthreads();                       // all waves done reading buf[cur^1]
    if (wid == 0) {
      if (it + 1 < steps) {
        tdm_load_tile(A + (size_t)bm * K + (it + 1) * 32, &As[cur ^ 1][0], K);
        tdm_load_tile(W + (size_t)bn * K + (it + 1) * 32, &Ws[cur ^ 1][0], K);
        __builtin_amdgcn_s_wait_tensorcnt(2);   // tile `it` (2 older ops) done
      } else {
        __builtin_amdgcn_s_wait_tensorcnt(0);
      }
    }
    __syncthreads();                       // tile `it` visible to all waves
#else
    const int cur = 0;
    const int k0 = it * 32;
    __syncthreads();
#pragma unroll
    for (int i = 0; i < 2; ++i) {          // A tile: 512 chunks of 8 bf16
      int c = tid + i * 256;
      int row = c >> 2, cc = c & 3;
      const bf16_t* gp = A + (size_t)(bm + row) * K + k0 + cc * 8;
      *(v8bf*)&As[0][row * LSTR + cc * 8] = *(const v8bf*)gp;
      if (k0 + 32 < K) __builtin_prefetch(gp + 32, 0, 1);
    }
#pragma unroll
    for (int i = 0; i < 2; ++i) {          // W tile: 512 chunks of 8 bf16
      int c = tid + i * 256;
      int row = c >> 2, cc = c & 3;
      const bf16_t* gp = W + (size_t)(bn + row) * K + k0 + cc * 8;
      *(v8bf*)&Ws[0][row * LSTR + cc * 8] = *(const v8bf*)gp;
      if (k0 + 32 < K) __builtin_prefetch(gp + 32, 0, 1);
    }
    __syncthreads();
#endif

    const bf16_t* Ab = &As[cur][0];
    const bf16_t* Wb = &Ws[cur][0];
    v16bf af[2], bfr[4];
    const int akb = hi ? 8 : 0;
#pragma unroll
    for (int mf = 0; mf < 2; ++mf) {
      int row = wm * 32 + mf * 16 + l15;
      FragU u;
      u.h[0] = *(const v8bf*)&Ab[row * LSTR + akb];
      u.h[1] = *(const v8bf*)&Ab[row * LSTR + akb + 16];
      af[mf] = u.v;
    }
    const int bkb = hi ? 8 : 0;
#pragma unroll
    for (int nf = 0; nf < 4; ++nf) {
      int row = wn * 64 + nf * 16 + l15;
      FragU u;
      u.h[0] = *(const v8bf*)&Wb[row * LSTR + bkb];
      u.h[1] = *(const v8bf*)&Wb[row * LSTR + bkb + 16];
      bfr[nf] = u.v;
    }
#pragma unroll
    for (int mf = 0; mf < 2; ++mf)
#pragma unroll
      for (int nf = 0; nf < 4; ++nf)
        acc[mf][nf] = __builtin_amdgcn_wmma_f32_16x16x32_bf16(
            false, af[mf], false, bfr[nf], (short)0, acc[mf][nf], false, false);
  }

  // Epilogue
#pragma unroll
  for (int mf = 0; mf < 2; ++mf) {
#pragma unroll
    for (int nf = 0; nf < 4; ++nf) {
      int col = bn + wn * 64 + nf * 16 + l15;
      float bv = (EPI >= 1) ? bias[col] : 0.0f;
#pragma unroll
      for (int v = 0; v < 8; ++v) {
        int row = bm + wm * 32 + mf * 16 + (hi ? v + 8 : v);
        float xv = acc[mf][nf][v] + bv;
        if (EPI == 2) xv += res[(size_t)row * N + col];
        if (EPI == 3) xv = xv / (1.0f + __expf(-1.702f * xv));
        if (OUT_BF16) ((bf16_t*)out)[(size_t)row * N + col] = (bf16_t)xv;
        else          ((float*)out)[(size_t)row * N + col] = xv;
      }
    }
  }
}

// ---------------------------------------------------------------------------
// Flash attention: block = (b*12+h, qblock of 128 rows); seq=1024, Dh=64
// ---------------------------------------------------------------------------
#define QSTR 72
#define VSTR 136
#define PSTR 136

__global__ __launch_bounds__(256)
void attn_kernel(const float* __restrict__ qkv, bf16_t* __restrict__ o) {
  __shared__ bf16_t Qs[128 * QSTR];
  __shared__ bf16_t Ks[128 * QSTR];
  __shared__ bf16_t Vt[64 * VSTR];
  __shared__ bf16_t Ps[8][16 * PSTR];

  const int bh = blockIdx.x, qb = blockIdx.y;
  const int bidx = bh / 12, hh = bh % 12;
  const int tid = threadIdx.x, wid = tid >> 5, lane = tid & 31, l15 = lane & 15;
  const bool hi = lane >= 16;
  const size_t baseQ = (size_t)bidx * 1024 * 2304 + hh * 64;
  const float* qp = qkv + baseQ;
  const float* kp = qkv + baseQ + 768;
  const float* vp = qkv + baseQ + 1536;

  // Stage Q (pre-scaled by 1/sqrt(64))
#pragma unroll
  for (int i = 0; i < 8; ++i) {
    int c = tid + i * 256;            // 2048 chunks of float4
    int row = c >> 4, ch = c & 15;
    float4 f = *(const float4*)(qp + (size_t)(qb * 128 + row) * 2304 + ch * 4);
    bf16_t* p = &Qs[row * QSTR + ch * 4];
    p[0] = (bf16_t)(f.x * 0.125f); p[1] = (bf16_t)(f.y * 0.125f);
    p[2] = (bf16_t)(f.z * 0.125f); p[3] = (bf16_t)(f.w * 0.125f);
  }

  v8f acc_o[4];
#pragma unroll
  for (int i = 0; i < 4; ++i) acc_o[i] = zero8();
  float m_i[8], l_i[8];
#pragma unroll
  for (int v = 0; v < 8; ++v) { m_i[v] = -INFINITY; l_i[v] = 0.0f; }

  for (int jb = 0; jb < 8; ++jb) {
    __syncthreads();
    // Stage K_j (row-major) and V_j (transposed to [dim][key])
#pragma unroll
    for (int i = 0; i < 8; ++i) {
      int c = tid + i * 256;
      int row = c >> 4, ch = c & 15;
      const size_t goff = (size_t)(jb * 128 + row) * 2304 + ch * 4;
      float4 f = *(const float4*)(kp + goff);
      bf16_t* p = &Ks[row * QSTR + ch * 4];
      p[0] = (bf16_t)f.x; p[1] = (bf16_t)f.y; p[2] = (bf16_t)f.z; p[3] = (bf16_t)f.w;
      float4 g = *(const float4*)(vp + goff);
      Vt[(ch * 4 + 0) * VSTR + row] = (bf16_t)g.x;
      Vt[(ch * 4 + 1) * VSTR + row] = (bf16_t)g.y;
      Vt[(ch * 4 + 2) * VSTR + row] = (bf16_t)g.z;
      Vt[(ch * 4 + 3) * VSTR + row] = (bf16_t)g.w;
    }
    __syncthreads();

    // S = (Q*scale) K^T : wave -> 16 q rows x 128 keys
    v8f sacc[8];
#pragma unroll
    for (int nf = 0; nf < 8; ++nf) sacc[nf] = zero8();
#pragma unroll
    for (int ks = 0; ks < 2; ++ks) {
      FragU ua;
      {
        int row = wid * 16 + l15;
        int kb = ks * 32 + (hi ? 8 : 0);
        ua.h[0] = *(const v8bf*)&Qs[row * QSTR + kb];
        ua.h[1] = *(const v8bf*)&Qs[row * QSTR + kb + 16];
      }
#pragma unroll
      for (int nf = 0; nf < 8; ++nf) {
        int krow = nf * 16 + l15;
        int kb = ks * 32 + (hi ? 8 : 0);
        FragU ub;
        ub.h[0] = *(const v8bf*)&Ks[krow * QSTR + kb];
        ub.h[1] = *(const v8bf*)&Ks[krow * QSTR + kb + 16];
        sacc[nf] = __builtin_amdgcn_wmma_f32_16x16x32_bf16(
            false, ua.v, false, ub.v, (short)0, sacc[nf], false, false);
      }
    }

    // Online softmax
    float mnew[8], alpha[8], rs[8];
#pragma unroll
    for (int v = 0; v < 8; ++v) {
      float t = sacc[0][v];
#pragma unroll
      for (int nf = 1; nf < 8; ++nf) t = fmaxf(t, sacc[nf][v]);
      t = wave_max16(t);
      mnew[v] = fmaxf(m_i[v], t);
      alpha[v] = __expf(m_i[v] - mnew[v]);
      rs[v] = 0.0f;
    }
#pragma unroll
    for (int nf = 0; nf < 8; ++nf)
#pragma unroll
      for (int v = 0; v < 8; ++v) {
        float pe = __expf(sacc[nf][v] - mnew[v]);
        rs[v] += pe;
        Ps[wid][(hi ? v + 8 : v) * PSTR + nf * 16 + l15] = (bf16_t)pe;
      }
#pragma unroll
    for (int v = 0; v < 8; ++v) {
      l_i[v] = l_i[v] * alpha[v] + wave_sum16(rs[v]);
      m_i[v] = mnew[v];
    }
#pragma unroll
    for (int nf2 = 0; nf2 < 4; ++nf2)
#pragma unroll
      for (int v = 0; v < 8; ++v) acc_o[nf2][v] *= alpha[v];

    // O += P @ V_j  (per-wave P tile re-laid out through LDS)
#pragma unroll
    for (int ks = 0; ks < 4; ++ks) {
      FragU ua;
      {
        int kb = ks * 32 + (hi ? 8 : 0);
        ua.h[0] = *(const v8bf*)&Ps[wid][l15 * PSTR + kb];
        ua.h[1] = *(const v8bf*)&Ps[wid][l15 * PSTR + kb + 16];
      }
#pragma unroll
      for (int nf2 = 0; nf2 < 4; ++nf2) {
        int drow = nf2 * 16 + l15;
        int kb = ks * 32 + (hi ? 8 : 0);
        FragU ub;
        ub.h[0] = *(const v8bf*)&Vt[drow * VSTR + kb];
        ub.h[1] = *(const v8bf*)&Vt[drow * VSTR + kb + 16];
        acc_o[nf2] = __builtin_amdgcn_wmma_f32_16x16x32_bf16(
            false, ua.v, false, ub.v, (short)0, acc_o[nf2], false, false);
      }
    }
  }

  // Write O / l  ->  o[b, s, h*64+d] (bf16)
  const size_t obase = (size_t)bidx * 1024 * 768 + hh * 64;
#pragma unroll
  for (int nf2 = 0; nf2 < 4; ++nf2)
#pragma unroll
    for (int v = 0; v < 8; ++v) {
      int qrow = qb * 128 + wid * 16 + (hi ? v + 8 : v);
      int d = nf2 * 16 + l15;
      o[obase + (size_t)qrow * 768 + d] = (bf16_t)(acc_o[nf2][v] / l_i[v]);
    }
}

// ---------------------------------------------------------------------------
extern "C" void kernel_launch(void* const* d_in, const int* in_sizes, int n_in,
                              void* d_out, int out_size, void* d_ws, size_t ws_size,
                              hipStream_t stream) {
  (void)in_sizes; (void)n_in; (void)out_size; (void)ws_size;
  const float* x         = (const float*)d_in[0];
  const float* in_proj_w = (const float*)d_in[1];
  const float* in_proj_b = (const float*)d_in[2];
  const float* lora_a    = (const float*)d_in[3];
  const float* lora_b    = (const float*)d_in[4];
  const float* out_w     = (const float*)d_in[5];
  const float* out_b     = (const float*)d_in[6];
  const float* fc_w      = (const float*)d_in[7];
  const float* fc_b      = (const float*)d_in[8];
  const float* proj_w    = (const float*)d_in[9];
  const float* proj_b    = (const float*)d_in[10];
  const float* ln1_g     = (const float*)d_in[11];
  const float* ln1_b     = (const float*)d_in[12];
  const float* ln2_g     = (const float*)d_in[13];
  const float* ln2_b     = (const float*)d_in[14];

  const int M = 8 * 1024;
  char* p = (char*)d_ws;
  float*  xcur = (float*)p;  p += (size_t)M * 768 * 4;
  bf16_t* hb   = (bf16_t*)p; p += (size_t)M * 768 * 2;
  float*  qkv  = (float*)p;  p += (size_t)M * 2304 * 4;
  bf16_t* obf  = (bf16_t*)p; p += (size_t)M * 768 * 2;
  bf16_t* mid  = (bf16_t*)p; p += (size_t)M * 3072 * 2;
  bf16_t* Wm   = (bf16_t*)p; p += (size_t)2304 * 768 * 2;
  bf16_t* owb  = (bf16_t*)p; p += (size_t)768 * 768 * 2;
  bf16_t* fcwb = (bf16_t*)p; p += (size_t)3072 * 768 * 2;
  bf16_t* prwb = (bf16_t*)p; p += (size_t)768 * 3072 * 2;

  (void)hipMemcpyAsync(xcur, x, (size_t)M * 768 * 4, hipMemcpyDeviceToDevice, stream);

  for (int L = 0; L < 4; ++L) {
    // Per-layer weight prep: LoRA fold (bf16) + f32->bf16 weight conversion
    merge_qkv_kernel<<<(2304 * 768) / 256, 256, 0, stream>>>(
        in_proj_w + (size_t)L * 2304 * 768, lora_a + (size_t)L * 8 * 768,
        lora_b + (size_t)L * 2304 * 8, Wm);
    cvt_bf16_kernel<<<(768 * 768) / 256, 256, 0, stream>>>(
        out_w + (size_t)L * 768 * 768, owb, 768 * 768);
    cvt_bf16_kernel<<<(3072 * 768) / 256, 256, 0, stream>>>(
        fc_w + (size_t)L * 3072 * 768, fcwb, 3072 * 768);
    cvt_bf16_kernel<<<(768 * 3072) / 256, 256, 0, stream>>>(
        proj_w + (size_t)L * 768 * 3072, prwb, 768 * 3072);

    ln_kernel<<<M, 256, 0, stream>>>(xcur, ln1_g + L * 768, ln1_b + L * 768, hb);
    gemm_kernel<1, false><<<dim3(2304 / 128, M / 128), 256, 0, stream>>>(
        hb, Wm, in_proj_b + (size_t)L * 2304, nullptr, qkv, M, 2304, 768);
    attn_kernel<<<dim3(96, 8), 256, 0, stream>>>(qkv, obf);
    gemm_kernel<2, false><<<dim3(768 / 128, M / 128), 256, 0, stream>>>(
        obf, owb, out_b + (size_t)L * 768, xcur, xcur, M, 768, 768);
    ln_kernel<<<M, 256, 0, stream>>>(xcur, ln2_g + L * 768, ln2_b + L * 768, hb);
    gemm_kernel<3, true><<<dim3(3072 / 128, M / 128), 256, 0, stream>>>(
        hb, fcwb, fc_b + (size_t)L * 3072, nullptr, mid, M, 3072, 768);
    gemm_kernel<2, false><<<dim3(768 / 128, M / 128), 256, 0, stream>>>(
        mid, prwb, proj_b + (size_t)L * 768, xcur, xcur, M, 768, 3072);
  }
  (void)hipMemcpyAsync(d_out, xcur, (size_t)M * 768 * 4, hipMemcpyDeviceToDevice, stream);
}
